// FeatureWiseAttentionBlock_78675210928177
// MI455X (gfx1250) — matrix operand order repaired and checked by reading, not compiled, not run on hardware
//
#include <hip/hip_runtime.h>
#include <stdint.h>

// Problem constants (match reference)
#define B_   32
#define L_   2048
#define E_   512
#define H_   512
#define NC   8            // L-chunks per batch -> 256 workgroups total
#define LC   (L_ / NC)    // 256 rows per chunk
#define TPB  256          // 8 wave32 waves
#define LDW  520          // padded LDS row stride (elements) -> conflict-free frag loads

typedef __attribute__((ext_vector_type(16))) __bf16 v16bf;
typedef __attribute__((ext_vector_type(8)))  float  v8f;

union FragAB { uint4 q[2]; v16bf v; };

__device__ __forceinline__ uint16_t f32_to_bf16(float f) {
    uint32_t u = __builtin_bit_cast(uint32_t, f);
    uint32_t r = u + 0x7FFFu + ((u >> 16) & 1u);   // round-to-nearest-even
    return (uint16_t)(r >> 16);
}
__device__ __forceinline__ float bf16_to_f32(uint16_t h) {
    uint32_t u = ((uint32_t)h) << 16;
    return __builtin_bit_cast(float, u);
}

// Branchless transcendentals on the v_exp_f32 / v_rcp_f32 hardware units.
__device__ __forceinline__ float fexp(float x) {          // exp(x), x <= 0 here
    return __builtin_amdgcn_exp2f(x * 1.44269504088896340736f);
}
__device__ __forceinline__ float ftanh(float x) {
    float xc = fminf(fmaxf(x, -15.f), 15.f);
    float t  = __builtin_amdgcn_exp2f(xc * 2.88539008177792681472f); // e^(2x)
    return (t - 1.f) * __builtin_amdgcn_rcpf(t + 1.f);
}

// ---------------- kernel 0: fp32 weights -> bf16 copies in workspace ----------
__global__ void prep_weights_k(const float* __restrict__ Wh, const float* __restrict__ Wc,
                               uint16_t* __restrict__ WhB, uint16_t* __restrict__ WcB) {
    int i = blockIdx.x * blockDim.x + threadIdx.x;
    if (i < E_ * H_) WhB[i] = f32_to_bf16(Wh[i]);
    int j = i - E_ * H_;
    if (j >= 0 && j < H_ * E_) WcB[j] = f32_to_bf16(Wc[j]);
}

// ---------------- kernel 1: fused GEMM1 -> tanh -> GEMM2 -> online softmax ----
__global__ __launch_bounds__(TPB, 2)
void fused_attn_k(const float* __restrict__ enc, const int* __restrict__ lengths,
                  const float* __restrict__ bh,
                  const uint16_t* __restrict__ WhB, const uint16_t* __restrict__ WcB,
                  float* __restrict__ pm, float* __restrict__ ps, float* __restrict__ pa) {
    __shared__ __align__(16) uint16_t encS[16 * LDW];  // 16-row enc tile, bf16
    __shared__ __align__(16) uint16_t hS[16 * LDW];    // 16-row hidden tile, bf16

    const int tid  = threadIdx.x;
    const int lane = tid & 31;
    const int wave = tid >> 5;
    const int half = lane >> 4;     // 0: rows 0-7 of C, 1: rows 8-15
    const int ln   = lane & 15;     // column within 16-wide tile / row of A

    const int b     = blockIdx.x / NC;
    const int chunk = blockIdx.x % NC;
    const int len   = lengths[b];
    const int base  = chunk * LC;
    const int pbase = (b * NC + chunk) * E_;

    if (base >= len) {              // empty chunk: neutral partials
        for (int e = tid; e < E_; e += TPB) {
            pm[pbase + e] = -__builtin_inff();
            ps[pbase + e] = 0.f;
            pa[pbase + e] = 0.f;
        }
        return;
    }

    // Bias value for each of this wave's 4 H-tiles (C layout: column = ln)
    float bhv[4];
#pragma unroll
    for (int t = 0; t < 4; ++t) bhv[t] = bh[wave * 64 + t * 16 + ln];

    // Online-softmax running state, one column per lane per E-tile (duplicated
    // across the two half-wave row groups; kept identical via shfl_xor merges).
    float m_run[4], s_run[4], a_run[4];
#pragma unroll
    for (int t = 0; t < 4; ++t) { m_run[t] = -__builtin_inff(); s_run[t] = 0.f; a_run[t] = 0.f; }

    int nrows = len - base; if (nrows > LC) nrows = LC;
    const int nblk = (nrows + 15) >> 4;

    const uint32_t aoff0 = ln * LDW + half * 8;        // A-frag LDS element offsets
    const uint32_t aoff1 = aoff0 + 16;

    for (int t = 0; t < nblk; ++t) {
        __syncthreads();  // previous block's readers of encS/hS are done

        // ---- cooperative load: 16 x 512 fp32 -> bf16 LDS (padded rows) ----
        const size_t gbase = (((size_t)b * L_) + base + t * 16) * E_;
#pragma unroll
        for (int it = 0; it < 8; ++it) {
            int i   = tid + it * TPB;       // 2048 float4s total
            int row = i >> 7;
            int c4  = (i & 127) << 2;
            float4 f = *(const float4*)(enc + gbase + (size_t)row * E_ + c4);
            uint32_t p0 = (uint32_t)f32_to_bf16(f.x) | ((uint32_t)f32_to_bf16(f.y) << 16);
            uint32_t p1 = (uint32_t)f32_to_bf16(f.z) | ((uint32_t)f32_to_bf16(f.w) << 16);
            *(uint2*)&encS[row * LDW + c4] = make_uint2(p0, p1);
        }
        if (t + 1 < nblk)  // CDNA5 global_prefetch of next tile
            __builtin_prefetch(enc + gbase + (size_t)16 * E_ + (size_t)tid * 32, 0, 0);
        __syncthreads();

        // ---- GEMM1: h = tanh(enc @ Wh + bh); k-outer, one A-frag feeds the
        //      wave's 4 H-tiles (32 accumulator VGPRs live, no spills) ----
        {
            v8f acc[4];
#pragma unroll
            for (int ht = 0; ht < 4; ++ht)
#pragma unroll
                for (int r = 0; r < 8; ++r) acc[ht][r] = bhv[ht];

#pragma clang loop unroll_count(2)
            for (int k = 0; k < E_; k += 32) {
                FragAB a;
                a.q[0] = *(const uint4*)&encS[aoff0 + k];
                a.q[1] = *(const uint4*)&encS[aoff1 + k];
                const uint16_t* wrow = WhB + (size_t)k * H_ + wave * 64 + ln;
#pragma unroll
                for (int ht = 0; ht < 4; ++ht) {
                    FragAB bf;
#pragma unroll
                    for (int s = 0; s < 16; ++s)
                        bf.v[s] = __builtin_bit_cast(__bf16, wrow[(size_t)(s < 8 ? 8 * half + s : 8 + 8 * half + s) * H_ + ht * 16]);
                    acc[ht] = __builtin_amdgcn_wmma_f32_16x16x32_bf16(
                        false, a.v, false, bf.v, (short)0, acc[ht], false, false);
                }
            }
#pragma unroll
            for (int ht = 0; ht < 4; ++ht) {
                const int h0 = wave * 64 + ht * 16;
#pragma unroll
                for (int r = 0; r < 8; ++r) {
                    int row = half * 8 + r;
                    hS[row * LDW + h0 + ln] = f32_to_bf16(ftanh(acc[ht][r]));
                }
            }
        }
        __syncthreads();

        // ---- GEMM2 (att = h @ Wc) + online softmax over this 16-row block ----
        int vrows = nrows - t * 16; if (vrows > 16) vrows = 16;
        {
            v8f acc[4];
#pragma unroll
            for (int et = 0; et < 4; ++et) acc[et] = (v8f){};

#pragma clang loop unroll_count(2)
            for (int k = 0; k < H_; k += 32) {
                FragAB a;
                a.q[0] = *(const uint4*)&hS[aoff0 + k];
                a.q[1] = *(const uint4*)&hS[aoff1 + k];
                const uint16_t* wrow = WcB + (size_t)k * E_ + wave * 64 + ln;
#pragma unroll
                for (int et = 0; et < 4; ++et) {
                    FragAB bf;
#pragma unroll
                    for (int s = 0; s < 16; ++s)
                        bf.v[s] = __builtin_bit_cast(__bf16, wrow[(size_t)(s < 8 ? 8 * half + s : 8 + 8 * half + s) * E_ + et * 16]);
                    acc[et] = __builtin_amdgcn_wmma_f32_16x16x32_bf16(
                        false, a.v, false, bf.v, (short)0, acc[et], false, false);
                }
            }

#pragma unroll
            for (int et = 0; et < 4; ++et) {
                const int e0 = wave * 64 + et * 16;
                // column e = e0 + ln; this lane holds rows half*8 .. half*8+7
                float vv[8];
                float lmax = -__builtin_inff();
#pragma unroll
                for (int r = 0; r < 8; ++r) {
                    int row = half * 8 + r;
                    vv[r] = (row < vrows) ? acc[et][r] : -__builtin_inff();
                    lmax = fmaxf(lmax, vv[r]);
                }
                float cmax = fmaxf(lmax, __shfl_xor(lmax, 16, 32));  // 16-row col max
                float newm = fmaxf(m_run[et], cmax);
                float scale = fexp(m_run[et] - newm);
                float psum = 0.f, pacc = 0.f;
#pragma unroll
                for (int r = 0; r < 8; ++r) {
                    int row = half * 8 + r;
                    float ev = (row < vrows) ? fexp(vv[r] - newm) : 0.f;
                    psum += ev;
                    pacc += ev * bf16_to_f32(encS[row * LDW + e0 + ln]);
                }
                psum += __shfl_xor(psum, 16, 32);
                pacc += __shfl_xor(pacc, 16, 32);
                s_run[et] = s_run[et] * scale + psum;
                a_run[et] = a_run[et] * scale + pacc;
                m_run[et] = newm;
            }
        }
    }

    if (half == 0) {   // lanes 0-15 publish chunk partials
#pragma unroll
        for (int et = 0; et < 4; ++et) {
            int e = wave * 64 + et * 16 + ln;
            pm[pbase + e] = m_run[et];
            ps[pbase + e] = s_run[et];
            pa[pbase + e] = a_run[et];
        }
    }
}

// ---------------- kernel 2: log-sum-exp merge of chunk partials ---------------
__global__ void combine_k(const float* __restrict__ pm, const float* __restrict__ ps,
                          const float* __restrict__ pa, float* __restrict__ out) {
    int i = blockIdx.x * blockDim.x + threadIdx.x;   // over B*E
    if (i >= B_ * E_) return;
    int b = i / E_, e = i % E_;
    float m = -__builtin_inff(), s = 0.f, a = 0.f;
    for (int c = 0; c < NC; ++c) {
        int idx = (b * NC + c) * E_ + e;
        float mc = pm[idx];
        if (!(mc > -__builtin_inff())) continue;     // empty chunk
        float sc = ps[idx], ac = pa[idx];
        if (!(m > -__builtin_inff())) { m = mc; s = sc; a = ac; }
        else {
            float nm = fmaxf(m, mc);
            float wa = fexp(m - nm), wb = fexp(mc - nm);
            s = s * wa + sc * wb;
            a = a * wa + ac * wb;
            m = nm;
        }
    }
    out[i] = a / s;
}

extern "C" void kernel_launch(void* const* d_in, const int* in_sizes, int n_in,
                              void* d_out, int out_size, void* d_ws, size_t ws_size,
                              hipStream_t stream) {
    const float* enc     = (const float*)d_in[0];   // [B,L,E] fp32
    const int*   lengths = (const int*)d_in[1];     // [B] int32
    const float* Wh      = (const float*)d_in[2];   // [E,H] fp32
    const float* bh      = (const float*)d_in[3];   // [H] fp32
    const float* Wc      = (const float*)d_in[4];   // [H,E] fp32
    float* out = (float*)d_out;                     // [B,1,E] fp32

    // workspace layout: Wh bf16 | Wc bf16 | partial m | partial s | partial acc
    uint16_t* WhB = (uint16_t*)d_ws;
    uint16_t* WcB = WhB + (size_t)E_ * H_;
    float* pm = (float*)((char*)d_ws + 2u * (size_t)E_ * H_ * sizeof(uint16_t));
    float* ps = pm + (size_t)B_ * NC * E_;
    float* pa = ps + (size_t)B_ * NC * E_;

    prep_weights_k<<<(2 * E_ * H_ + TPB - 1) / TPB, TPB, 0, stream>>>(Wh, Wc, WhB, WcB);
    fused_attn_k<<<B_ * NC, TPB, 0, stream>>>(enc, lengths, bh, WhB, WcB, pm, ps, pa);
    combine_k<<<(B_ * E_ + TPB - 1) / TPB, TPB, 0, stream>>>(pm, ps, pa, out);
}
